// MLP_Energy_Head_31928786878751
// MI455X (gfx1250) — matrix operand order repaired and checked by reading, not compiled, not run on hardware
//
#include <hip/hip_runtime.h>

#define NNODES 131072
#define NLSQ   9
#define CDIM   128
#define HDIM   1024
#define NMOLS  2048
#define TILE_M 64
#define XS     136   // padded LDS row stride for X bf16 (elements)
#define XSF    132   // padded LDS row stride for X fp32 staging (elements)
#define HS     1040  // padded LDS row stride for H1 bf16 (elements)

#define SH1_BYTES (TILE_M * HS * 2)          // 133120
#define SX_BYTES  (TILE_M * XS * 2)          //  17408
#define SE_BYTES  (TILE_M * 4)               //    256

typedef __attribute__((ext_vector_type(16))) __bf16       v16bf;
typedef __attribute__((ext_vector_type(8)))  float        v8f;
typedef __attribute__((ext_vector_type(4)))  unsigned int v4u;
typedef __attribute__((ext_vector_type(4)))  int          v4i;

// b128 async-copy pointer types: <4 x i32> in global (AS1) / LDS (AS3)
typedef __attribute__((address_space(1))) v4i* gb128_ptr;
typedef __attribute__((address_space(3))) v4i* lb128_ptr;

#if defined(__gfx1250__) && __has_builtin(__builtin_amdgcn_global_load_async_to_lds_b128)
#define HAVE_ASYNC_LDS 1
#else
#define HAVE_ASYNC_LDS 0
#endif

union BF16Frag { v16bf v; v4u q[2]; };

__device__ __forceinline__ unsigned short f2bf(float f) {
  unsigned int u = __float_as_uint(f);
  u += 0x7FFFu + ((u >> 16) & 1u);   // round-to-nearest-even
  return (unsigned short)(u >> 16);
}

__device__ __forceinline__ float silu_f(float x) {
  return x / (1.0f + __expf(-x));
}

#if HAVE_ASYNC_LDS
__device__ __forceinline__ void wait_async0() {
#if __has_builtin(__builtin_amdgcn_s_wait_asynccnt)
  __builtin_amdgcn_s_wait_asynccnt(0);
#else
  asm volatile("s_wait_asynccnt 0" ::: "memory");
#endif
}
#endif

// Transpose + fp32->bf16: W (K x N row-major) -> Wt (N x K row-major)
__global__ void wt_convert_kernel(const float* __restrict__ W,
                                  unsigned short* __restrict__ Wt,
                                  int K, int N) {
  const int total = K * N;
  for (int i = blockIdx.x * blockDim.x + threadIdx.x; i < total;
       i += gridDim.x * blockDim.x) {
    const int n = i / K;
    const int k = i - n * K;
    Wt[i] = f2bf(W[k * N + n]);
  }
}

__global__ void zero_out_kernel(float* __restrict__ out, int n) {
  const int i = blockIdx.x * blockDim.x + threadIdx.x;
  if (i < n) out[i] = 0.0f;
}

__global__ __launch_bounds__(256) void mlp_energy_kernel(
    const float* __restrict__ NE,            // [N, 9, 128] fp32
    const unsigned short* __restrict__ W1t,  // bf16 [1024 x 128]  (W1^T)
    const float* __restrict__ b1,            // [1024]
    const unsigned short* __restrict__ W2t,  // bf16 [1024 x 1024] (W2^T)
    const float* __restrict__ b2,            // [1024]
    const float* __restrict__ W3,            // [1024] fp32
    const float* __restrict__ b3,            // [1] fp32
    const int*   __restrict__ batch,         // [N]
    float* __restrict__ out)                 // [2048]
{
  // One LDS arena; the fp32 X-staging buffer aliases sH1 (dead before GEMM1
  // writes sH1; separated by __syncthreads).
  __shared__ __align__(16) unsigned char smem[SH1_BYTES + SX_BYTES + SE_BYTES];
  unsigned short* sH1 = (unsigned short*)smem;                  // bf16 [64 x HS]
  float*          sXf = (float*)smem;                           // fp32 [64 x XSF] (aliases sH1)
  unsigned short* sX  = (unsigned short*)(smem + SH1_BYTES);    // bf16 [64 x XS]
  float*          sE  = (float*)(smem + SH1_BYTES + SX_BYTES);  // [64]

  const int tid    = threadIdx.x;
  const int wave   = tid >> 5;
  const int lane   = tid & 31;
  const int lane15 = lane & 15;
  const int hi     = lane >> 4;
  const int g0     = blockIdx.x * TILE_M;

#if HAVE_ASYNC_LDS
  // ---- stage X tile (fp32) straight into LDS via the async engine ----
  for (int i = tid; i < TILE_M * (CDIM / 4); i += 256) {
    const int row = i >> 5;            // 32 x b128 chunks per row
    const int c4  = (i & 31) << 2;
    const float* gp = NE + (size_t)(g0 + row) * (NLSQ * CDIM) + c4;
    float*       lp = &sXf[row * XSF + c4];
    __builtin_amdgcn_global_load_async_to_lds_b128(
        (gb128_ptr)gp, (lb128_ptr)lp, 0, 0);
  }
  if (tid < TILE_M) sE[tid] = 0.0f;
  wait_async0();
  __syncthreads();
  // ---- convert staged fp32 -> bf16 (LDS -> LDS) ----
  for (int i = tid; i < TILE_M * (CDIM / 4); i += 256) {
    const int row = i >> 5;
    const int c4  = (i & 31) << 2;
    const float4 f = *(const float4*)&sXf[row * XSF + c4];
    const unsigned int lo = (unsigned int)f2bf(f.x) | ((unsigned int)f2bf(f.y) << 16);
    const unsigned int hw = (unsigned int)f2bf(f.z) | ((unsigned int)f2bf(f.w) << 16);
    unsigned int* p = (unsigned int*)&sX[row * XS + c4];
    p[0] = lo;
    p[1] = hw;
  }
  __syncthreads();
#else
  // ---- fallback: direct global load + convert ----
  for (int i = tid; i < TILE_M * (CDIM / 4); i += 256) {
    const int row = i >> 5;
    const int c4  = (i & 31) << 2;
    const float4 f =
        *(const float4*)(NE + (size_t)(g0 + row) * (NLSQ * CDIM) + c4);
    const unsigned int lo = (unsigned int)f2bf(f.x) | ((unsigned int)f2bf(f.y) << 16);
    const unsigned int hw = (unsigned int)f2bf(f.z) | ((unsigned int)f2bf(f.w) << 16);
    unsigned int* p = (unsigned int*)&sX[row * XS + c4];
    p[0] = lo;
    p[1] = hw;
  }
  if (tid < TILE_M) sE[tid] = 0.0f;
  __syncthreads();
#endif

  // ---- GEMM1: H1 = silu(X @ W1 + b1), M=64 K=128 N=1024 ----
  {
    // Manually hoist all 16 A fragments (4 m-tiles x 4 K-steps) into VGPRs;
    // they are invariant across the 8 n-sub-tiles.
    BF16Frag aF[16];
#pragma unroll
    for (int kk = 0; kk < 4; ++kk) {
#pragma unroll
      for (int m = 0; m < 4; ++m) {
        const int row = m * 16 + lane15;
        const unsigned short* ap = &sX[row * XS + kk * 32 + hi * 16];
        aF[kk * 4 + m].q[0] = *(const v4u*)(ap);
        aF[kk * 4 + m].q[1] = *(const v4u*)(ap + 8);
      }
    }
    const int ncol0 = wave * 128;
    for (int ns = 0; ns < 8; ++ns) {
      const int n0  = ncol0 + ns * 16;
      const int col = n0 + lane15;
      v8f acc[4] = {};
      const unsigned short* bbase = W1t + col * CDIM;
#pragma unroll
      for (int kk = 0; kk < 4; ++kk) {
        BF16Frag b;
        const unsigned short* bp = bbase + kk * 32 + hi * 16;
        b.q[0] = *(const v4u*)(bp);
        b.q[1] = *(const v4u*)(bp + 8);
#pragma unroll
        for (int m = 0; m < 4; ++m) {
          acc[m] = __builtin_amdgcn_wmma_f32_16x16x32_bf16(
              false, aF[kk * 4 + m].v, false, b.v, (short)0, acc[m], false, false);
        }
      }
      const float bias = b1[col];
#pragma unroll
      for (int m = 0; m < 4; ++m) {
#pragma unroll
        for (int v = 0; v < 8; ++v) {
          const int row = m * 16 + v + hi * 8;
          const float h = acc[m][v] + bias;
          sH1[row * HS + col] = f2bf(silu_f(h));
        }
      }
    }
  }
  __syncthreads();

  // ---- GEMM2 + head: e_row += sum_col silu(H1@W2+b2)_col * W3[col] ----
  // Block 2 n-sub-tiles per pass: each A fragment feeds 2 WMMAs, each B
  // fragment feeds 4 -> halves LDS read amplification on the A operand.
  {
    const int ncol0 = wave * 128;
    for (int ns = 0; ns < 8; ns += 2) {
      const int col0 = ncol0 + ns * 16 + lane15;
      const int col1 = col0 + 16;
      v8f acc0[4] = {};
      v8f acc1[4] = {};
      const unsigned short* bbase0 = W2t + col0 * HDIM;
      const unsigned short* bbase1 = W2t + col1 * HDIM;
      for (int k0 = 0; k0 < HDIM; k0 += 32) {
        // L2 prefetch a few K-steps ahead on both column streams
        __builtin_prefetch(bbase0 + k0 + 128, 0, 0);
        __builtin_prefetch(bbase1 + k0 + 128, 0, 0);
        BF16Frag b0, b1f;
        const unsigned short* bp0 = bbase0 + k0 + hi * 16;
        const unsigned short* bp1 = bbase1 + k0 + hi * 16;
        b0.q[0]  = *(const v4u*)(bp0);
        b0.q[1]  = *(const v4u*)(bp0 + 8);
        b1f.q[0] = *(const v4u*)(bp1);
        b1f.q[1] = *(const v4u*)(bp1 + 8);
#pragma unroll
        for (int m = 0; m < 4; ++m) {
          const int row = m * 16 + lane15;
          const unsigned short* ap = &sH1[row * HS + k0 + hi * 16];
          BF16Frag a;
          a.q[0] = *(const v4u*)(ap);
          a.q[1] = *(const v4u*)(ap + 8);
          acc0[m] = __builtin_amdgcn_wmma_f32_16x16x32_bf16(
              false, a.v, false, b0.v, (short)0, acc0[m], false, false);
          acc1[m] = __builtin_amdgcn_wmma_f32_16x16x32_bf16(
              false, a.v, false, b1f.v, (short)0, acc1[m], false, false);
        }
      }
      const float bias0 = b2[col0];
      const float w3c0  = W3[col0];
      const float bias1 = b2[col1];
      const float w3c1  = W3[col1];
#pragma unroll
      for (int m = 0; m < 4; ++m) {
#pragma unroll
        for (int v = 0; v < 8; ++v) {
          const int row = m * 16 + v + hi * 8;
          const float h0 = acc0[m][v] + bias0;
          const float h1 = acc1[m][v] + bias1;
          float t = silu_f(h0) * w3c0 + silu_f(h1) * w3c1;
          t += __shfl_xor(t, 8, 16);
          t += __shfl_xor(t, 4, 16);
          t += __shfl_xor(t, 2, 16);
          t += __shfl_xor(t, 1, 16);
          if (lane15 == 0) atomicAdd(&sE[row], t);
        }
      }
    }
  }
  __syncthreads();

  // ---- segment sum: one global float atomic per node ----
  if (tid < TILE_M) {
    const int g = g0 + tid;
    if (g < NNODES) {
      const float e = sE[tid] + b3[0];
      atomicAdd(&out[batch[g]], e);
    }
  }
}

extern "C" void kernel_launch(void* const* d_in, const int* in_sizes, int n_in,
                              void* d_out, int out_size, void* d_ws, size_t ws_size,
                              hipStream_t stream) {
  (void)in_sizes; (void)n_in; (void)out_size; (void)ws_size;
  const float* NE    = (const float*)d_in[0];
  const float* W1    = (const float*)d_in[1];
  const float* b1    = (const float*)d_in[2];
  const float* W2    = (const float*)d_in[3];
  const float* b2    = (const float*)d_in[4];
  const float* W3    = (const float*)d_in[5];
  const float* b3    = (const float*)d_in[6];
  const int*   batch = (const int*)d_in[7];
  float* out = (float*)d_out;

  unsigned short* W1t = (unsigned short*)d_ws;                       // 256 KB
  unsigned short* W2t = (unsigned short*)((char*)d_ws + 256 * 1024); // 2 MB

  zero_out_kernel<<<(NMOLS + 255) / 256, 256, 0, stream>>>(out, NMOLS);
  wt_convert_kernel<<<512, 256, 0, stream>>>(W1, W1t, CDIM, HDIM);
  wt_convert_kernel<<<2048, 256, 0, stream>>>(W2, W2t, HDIM, HDIM);
  mlp_energy_kernel<<<NNODES / TILE_M, 256, 0, stream>>>(
      NE, W1t, b1, W2t, b2, W3, b3, batch, out);
}
